// TopoPool_v3_13460427505956
// MI455X (gfx1250) — matrix operand order repaired and checked by reading, not compiled, not run on hardware
//
#include <hip/hip_runtime.h>
#include <hip/hip_bf16.h>
#include <math.h>

// ---------------------------------------------------------------------------
// TopoPool for MI455X (gfx1250, wave32).
//  - elev = x@W+b via v_wmma_f32_16x16x32_f16 (W broadcast into all 16 B cols)
//  - edge stats / flood-fill / segmented softmax pooling via global atomics
//  - fixed 64 flood iterations with per-iteration convergence flag (graph-safe)
// ---------------------------------------------------------------------------

typedef __attribute__((ext_vector_type(16))) _Float16 v16h;
typedef __attribute__((ext_vector_type(8)))  float    v8f;

#define TOPO_ITERS 64
#define TOPO_C 128

// ---- sortable-key float <-> uint for atomicMax over signed floats ----------
__device__ __forceinline__ unsigned topo_fkey(float f) {
  unsigned u = __float_as_uint(f);
  return (u & 0x80000000u) ? ~u : (u | 0x80000000u);
}
__device__ __forceinline__ float topo_funkey(unsigned k) {
  unsigned u = (k & 0x80000000u) ? (k & 0x7FFFFFFFu) : ~k;
  return __uint_as_float(u);
}

// ---------------------------------------------------------------------------
// Workspace / output init
// ---------------------------------------------------------------------------
__global__ void topo_ws_init_kernel(int* deg, int* gts, int* lts, int* best,
                                    unsigned* smax, float* zsum, int* changed,
                                    int N, int SENT) {
  int i = blockIdx.x * blockDim.x + threadIdx.x;
  if (i < N) {
    deg[i] = 0; gts[i] = 0; lts[i] = 0;
    best[i] = SENT;
    smax[i] = 0u;          // below key(-inf) for all real floats
    zsum[i] = 0.0f;
  }
  if (i < TOPO_ITERS) changed[i] = 0;
}

__global__ void topo_out_init_kernel(float* pooled, float* out_cb, int N) {
  long long t = (long long)blockIdx.x * blockDim.x + threadIdx.x;
  long long total = (long long)N * TOPO_C;
  if (t < total) pooled[t] = 0.0f;
  if (t < N)     out_cb[t] = -1.0f;
}

// ---------------------------------------------------------------------------
// elev = x @ W + b  via WMMA. One wave handles 16 nodes; W is broadcast into
// all 16 columns of B so every output column equals elev (we read column 0).
// No early returns: EXEC must be all-ones for v_wmma.
// ---------------------------------------------------------------------------
__global__ void topo_elev_wmma_kernel(const float* __restrict__ x,
                                      const float* __restrict__ W,
                                      const float* __restrict__ bptr,
                                      float* __restrict__ elev, int N) {
  const int lane = threadIdx.x & 31;
  const int wave = threadIdx.x >> 5;
  const int base = (blockIdx.x * 8 + wave) * 16;       // 16 rows per wave
  int row = base + (lane & 15);
  int rowc = row < N ? row : (N - 1);                  // clamp loads
  const int khalfA = (lane < 16) ? 0 : 16;             // A: K sub-range per lane half
  const int kbaseB = (lane < 16) ? 0 : 16;             // B: K sub-range per lane half

  v8f c = {};
  const float* xr = x + (size_t)rowc * TOPO_C;
  for (int kb = 0; kb < TOPO_C; kb += 32) {
    v16h a, bm;
#pragma unroll
    for (int e = 0; e < 8; ++e) a[e]     = (_Float16)xr[kb + khalfA + e];
#pragma unroll
    for (int e = 0; e < 8; ++e) a[8 + e] = (_Float16)xr[kb + 8 + khalfA + e];
#pragma unroll
    for (int e = 0; e < 16; ++e) bm[e]   = (_Float16)W[kb + kbaseB + e];
    // v_wmma_f32_16x16x32_f16 (8 args: neg_a, A, neg_b, B, c_mod, C, reuse_a, reuse_b)
    c = __builtin_amdgcn_wmma_f32_16x16x32_f16(false, a, false, bm,
                                               (short)0, c, false, false);
  }
  const float bias = bptr[0];
  // C/D layout: lane 0 holds (M=0..7, N=0) in c[0..7]; lane 16 holds M=8..15.
  if (lane == 0) {
#pragma unroll
    for (int r = 0; r < 8; ++r)
      if (base + r < N) elev[base + r] = c[r] + bias;
  } else if (lane == 16) {
#pragma unroll
    for (int r = 0; r < 8; ++r)
      if (base + 8 + r < N) elev[base + 8 + r] = c[r] + bias;
  }
}

// ---------------------------------------------------------------------------
// Per-edge degree / peak-trough counting
// ---------------------------------------------------------------------------
__global__ void topo_edge_stats_kernel(const int* __restrict__ src,
                                       const int* __restrict__ dst,
                                       const float* __restrict__ elev,
                                       int* deg, int* gts, int* lts, int E) {
  int e = blockIdx.x * blockDim.x + threadIdx.x;
  if (e >= E) return;
  int s = src[e], d = dst[e];
  float es = elev[s], ed = elev[d];
  atomicAdd(&deg[d], 1);
  if (ed >= es) atomicAdd(&gts[d], 1);
  if (ed <= es) atomicAdd(&lts[d], 1);
}

__global__ void topo_peak_kernel(const int* __restrict__ deg,
                                 const int* __restrict__ gts,
                                 const int* __restrict__ lts,
                                 int* assign, float* out_is_peak,
                                 float* out_is_trough, int N) {
  int i = blockIdx.x * blockDim.x + threadIdx.x;
  if (i >= N) return;
  bool pk = (gts[i] == deg[i]);
  bool tr = (lts[i] == deg[i]);
  out_is_peak[i]   = pk ? 1.0f : 0.0f;
  out_is_trough[i] = tr ? 1.0f : 0.0f;
  assign[i] = pk ? i : -1;
}

// ---------------------------------------------------------------------------
// Watershed flood-fill: fixed iteration count, per-iteration converged flag.
// ---------------------------------------------------------------------------
__global__ void topo_flood_edge_kernel(const int* __restrict__ src,
                                       const int* __restrict__ dst,
                                       const float* __restrict__ elev,
                                       const int* __restrict__ assign,
                                       int* best, const int* __restrict__ changed,
                                       int it, int E) {
  if (it > 0 && changed[it - 1] == 0) return;   // converged: cheap early exit
  int e = blockIdx.x * blockDim.x + threadIdx.x;
  if (e >= E) return;
  int s = src[e], d = dst[e];
  if (elev[d] <= elev[s]) {                     // "down" edge
    int as_ = assign[s];
    if (as_ >= 0 && assign[d] < 0) atomicMin(&best[d], as_);
  }
}

__global__ void topo_flood_node_kernel(int* assign, int* best, int* changed,
                                       int it, int N, int SENT) {
  if (it > 0 && changed[it - 1] == 0) return;
  int i = blockIdx.x * blockDim.x + threadIdx.x;
  if (i >= N) return;
  int bv = best[i];
  best[i] = SENT;                               // reset for next iteration
  if (assign[i] < 0 && bv < SENT) {
    assign[i] = bv;
    changed[it] = 1;                            // racy OR, fine
  }
}

// ---------------------------------------------------------------------------
// new_id = cumsum(is_peak) - 1 : 3-pass block scan (nblocks <= 512 for N=100K)
// ---------------------------------------------------------------------------
__global__ void topo_block_sum_kernel(const int* __restrict__ deg,
                                      const int* __restrict__ gts,
                                      int* bsum, int N) {
  __shared__ int sh[256];
  int i = blockIdx.x * 256 + threadIdx.x;
  sh[threadIdx.x] = (i < N && gts[i] == deg[i]) ? 1 : 0;
  __syncthreads();
  for (int s = 128; s > 0; s >>= 1) {
    if ((int)threadIdx.x < s) sh[threadIdx.x] += sh[threadIdx.x + s];
    __syncthreads();
  }
  if (threadIdx.x == 0) bsum[blockIdx.x] = sh[0];
}

__global__ void topo_scan_bsum_kernel(int* bsum, int nb) {
  __shared__ int sh[512];
  int t = threadIdx.x;
  sh[t] = (t < nb) ? bsum[t] : 0;
  __syncthreads();
  for (int off = 1; off < 512; off <<= 1) {
    int v = (t >= off) ? sh[t - off] : 0;
    __syncthreads();
    sh[t] += v;
    __syncthreads();
  }
  if (t < nb) bsum[t] = sh[t];                  // inclusive block-prefix sums
}

__global__ void topo_newid_kernel(const int* __restrict__ deg,
                                  const int* __restrict__ gts,
                                  const int* __restrict__ bsum,
                                  int* newid, int N) {
  __shared__ int sh[256];
  int t = threadIdx.x;
  int i = blockIdx.x * 256 + t;
  sh[t] = (i < N && gts[i] == deg[i]) ? 1 : 0;
  __syncthreads();
  for (int off = 1; off < 256; off <<= 1) {     // Hillis-Steele inclusive
    int v = (t >= off) ? sh[t - off] : 0;
    __syncthreads();
    sh[t] += v;
    __syncthreads();
  }
  int boff = (blockIdx.x > 0) ? bsum[blockIdx.x - 1] : 0;
  if (i < N) newid[i] = boff + sh[t] - 1;       // cumsum - 1
}

// ---------------------------------------------------------------------------
// seg id per node + cluster_batch scatter
// ---------------------------------------------------------------------------
__global__ void topo_seg_kernel(const int* __restrict__ assign,
                                const int* __restrict__ newid,
                                const int* __restrict__ deg,
                                const int* __restrict__ gts,
                                const int* __restrict__ batch,
                                int* segi, float* out_seg, float* out_cb, int N) {
  int i = blockIdx.x * blockDim.x + threadIdx.x;
  if (i >= N) return;
  int a = assign[i];
  int s = (a >= 0) ? newid[a] : 0;
  segi[i] = s;
  out_seg[i] = (float)s;
  if (gts[i] == deg[i]) out_cb[newid[i]] = (float)batch[i];   // peak node
}

// ---------------------------------------------------------------------------
// Segmented softmax + weighted pooling
// ---------------------------------------------------------------------------
__global__ void topo_segmax_kernel(const float* __restrict__ elev,
                                   const int* __restrict__ assign,
                                   const int* __restrict__ segi,
                                   unsigned* smax, int N) {
  int i = blockIdx.x * blockDim.x + threadIdx.x;
  if (i >= N) return;
  float v = (assign[i] >= 0) ? elev[i] : -1e30f;
  atomicMax(&smax[segi[i]], topo_fkey(v));
}

__global__ void topo_z_kernel(const float* __restrict__ elev,
                              const int* __restrict__ assign,
                              const int* __restrict__ segi,
                              const unsigned* __restrict__ smax,
                              float* wgt, float* zsum, int N) {
  int i = blockIdx.x * blockDim.x + threadIdx.x;
  if (i >= N) return;
  int s = segi[i];
  float w = 0.0f;
  if (assign[i] >= 0) w = expf(elev[i] - topo_funkey(smax[s]));
  wgt[i] = w;
  if (w != 0.0f) atomicAdd(&zsum[s], w);
}

__global__ void topo_scaling_kernel(const float* __restrict__ wgt,
                                    const float* __restrict__ zsum,
                                    const int* __restrict__ segi,
                                    float* scal, float* out_scaling, int N) {
  int i = blockIdx.x * blockDim.x + threadIdx.x;
  if (i >= N) return;
  float s = wgt[i] / (zsum[segi[i]] + 1e-12f);
  scal[i] = s;
  out_scaling[i] = s;
}

__global__ void topo_pool_kernel(const float* __restrict__ x,
                                 const float* __restrict__ scal,
                                 const int* __restrict__ segi,
                                 float* pooled, int N) {
  long long t = (long long)blockIdx.x * blockDim.x + threadIdx.x;
  long long total = (long long)N * TOPO_C;
  if (t >= total) return;
  int node = (int)(t >> 7);                     // /128
  int ch   = (int)(t & (TOPO_C - 1));
  float v = x[t] * scal[node];
  atomicAdd(&pooled[(size_t)segi[node] * TOPO_C + ch], v);
}

// ---------------------------------------------------------------------------
// Host launcher
// ---------------------------------------------------------------------------
extern "C" void kernel_launch(void* const* d_in, const int* in_sizes, int n_in,
                              void* d_out, int out_size, void* d_ws, size_t ws_size,
                              hipStream_t stream) {
  const float* x     = (const float*)d_in[0];   // [N,128]
  const float* W     = (const float*)d_in[1];   // [128,1]
  const float* bias  = (const float*)d_in[2];   // [1]
  const int*   eidx  = (const int*)d_in[3];     // [2, 2E] flat
  const int*   batch = (const int*)d_in[4];     // [N]

  const int N = in_sizes[4];
  const int E = in_sizes[3] / 2;                // directed edges (symmetrized)
  const int SENT = N;
  const int* src = eidx;
  const int* dst = eidx + E;

  // outputs: pooled[N*128] | seg[N] | scaling[N] | cluster_batch[N] | is_peak[N] | is_trough[N]
  float* out          = (float*)d_out;
  float* out_pooled   = out;
  float* out_seg      = out + (size_t)N * TOPO_C;
  float* out_scaling  = out_seg + N;
  float* out_cb       = out_scaling + N;
  float* out_is_peak  = out_cb + N;
  float* out_is_trough= out_is_peak + N;

  // workspace carve-out
  char* wsp = (char*)d_ws;
  size_t off = 0;
  auto alloc = [&](size_t bytes) -> void* {
    void* p = wsp + off;
    off = (off + bytes + 255) & ~(size_t)255;
    return p;
  };
  float*    elev   = (float*)alloc((size_t)N * 4);
  int*      deg    = (int*)alloc((size_t)N * 4);
  int*      gts    = (int*)alloc((size_t)N * 4);
  int*      lts    = (int*)alloc((size_t)N * 4);
  int*      assign = (int*)alloc((size_t)N * 4);
  int*      best   = (int*)alloc((size_t)N * 4);
  int*      newid  = (int*)alloc((size_t)N * 4);
  int*      segi   = (int*)alloc((size_t)N * 4);
  unsigned* smax   = (unsigned*)alloc((size_t)N * 4);
  float*    zsum   = (float*)alloc((size_t)N * 4);
  float*    wgt    = (float*)alloc((size_t)N * 4);
  float*    scal   = (float*)alloc((size_t)N * 4);
  int*      changed= (int*)alloc((size_t)TOPO_ITERS * 4);
  const int nb = (N + 255) / 256;
  int*      bsum   = (int*)alloc((size_t)nb * 4);

  const int TB = 256;
  const int gN  = (N + TB - 1) / TB;
  const int gE  = (E + TB - 1) / TB;
  const long long NC = (long long)N * TOPO_C;
  const int gNC = (int)((NC + TB - 1) / TB);

  // init
  topo_ws_init_kernel<<<gN, TB, 0, stream>>>(deg, gts, lts, best, smax, zsum,
                                             changed, N, SENT);
  topo_out_init_kernel<<<gNC, TB, 0, stream>>>(out_pooled, out_cb, N);

  // elev via WMMA: 8 waves/block, 16 rows/wave -> 128 rows/block
  const int gElev = (N + 127) / 128;
  topo_elev_wmma_kernel<<<gElev, 256, 0, stream>>>(x, W, bias, elev, N);

  // peak/trough detection
  topo_edge_stats_kernel<<<gE, TB, 0, stream>>>(src, dst, elev, deg, gts, lts, E);
  topo_peak_kernel<<<gN, TB, 0, stream>>>(deg, gts, lts, assign,
                                          out_is_peak, out_is_trough, N);

  // watershed flood fill (fixed launches, flag-gated)
  for (int it = 0; it < TOPO_ITERS; ++it) {
    topo_flood_edge_kernel<<<gE, TB, 0, stream>>>(src, dst, elev, assign, best,
                                                  changed, it, E);
    topo_flood_node_kernel<<<gN, TB, 0, stream>>>(assign, best, changed, it, N, SENT);
  }

  // relabel peaks -> contiguous cluster ids
  topo_block_sum_kernel<<<nb, 256, 0, stream>>>(deg, gts, bsum, N);
  topo_scan_bsum_kernel<<<1, 512, 0, stream>>>(bsum, nb);
  topo_newid_kernel<<<nb, 256, 0, stream>>>(deg, gts, bsum, newid, N);
  topo_seg_kernel<<<gN, TB, 0, stream>>>(assign, newid, deg, gts, batch,
                                         segi, out_seg, out_cb, N);

  // segmented softmax + pooling
  topo_segmax_kernel<<<gN, TB, 0, stream>>>(elev, assign, segi, smax, N);
  topo_z_kernel<<<gN, TB, 0, stream>>>(elev, assign, segi, smax, wgt, zsum, N);
  topo_scaling_kernel<<<gN, TB, 0, stream>>>(wgt, zsum, segi, scal, out_scaling, N);
  topo_pool_kernel<<<gNC, TB, 0, stream>>>(x, scal, segi, out_pooled, N);
}
